// Attention_15418932593462
// MI455X (gfx1250) — matrix operand order repaired and checked by reading, not compiled, not run on hardware
//
#include <hip/hip_runtime.h>

typedef __bf16 bf16;
typedef __attribute__((ext_vector_type(16))) __bf16 v16bf;
typedef __attribute__((ext_vector_type(8)))  __bf16 v8bf;
typedef __attribute__((ext_vector_type(2)))  __bf16 v2bf;
typedef __attribute__((ext_vector_type(8)))  float  v8f;

// Problem constants (from reference): B=2, N=4096, C=512, H=8, D=64
#define BB 2
#define NN 4096
#define CC 512
#define HH 8
#define DD 64

__device__ __forceinline__ v8f wmma_bf16(v16bf a, v16bf b, v8f c) {
    // D = A(16x32 bf16) x B(32x16 bf16) + C(16x16 f32)
    return __builtin_amdgcn_wmma_f32_16x16x32_bf16(
        /*neg_a=*/false, a, /*neg_b=*/false, b,
        /*c_mod=*/(short)0, c, /*reuse_a=*/false, /*reuse_b=*/false);
}

// A-fragment 16x32: lane(l): row = l&15, half = l>>4, K = {8h..8h+7} U {16+8h..23+8h}
__device__ __forceinline__ v16bf load_a_frag(const bf16* base, int ld) {
    const int lane = threadIdx.x & 31;
    const int row  = lane & 15;
    const int hlf  = lane >> 4;
    const bf16* p = base + row * ld + 8 * hlf;
    v8bf lo = *(const v8bf*)(p);
    v8bf hi = *(const v8bf*)(p + 16);
    return __builtin_shufflevector(lo, hi, 0,1,2,3,4,5,6,7,8,9,10,11,12,13,14,15);
}

// B-fragment 32x16 from K-contiguous-per-column storage (elem (k,n) at base[n*ld + k])
// lane(l): col = l&15, half = l>>4, K = {8h..8h+7} U {16+8h..23+8h}
__device__ __forceinline__ v16bf load_b_frag(const bf16* base, int ld) {
    const int lane = threadIdx.x & 31;
    const int col  = lane & 15;
    const int hlf  = lane >> 4;
    const bf16* p = base + col * ld + 8 * hlf;
    v8bf lo = *(const v8bf*)(p);
    v8bf hi = *(const v8bf*)(p + 16);
    return __builtin_shufflevector(lo, hi, 0,1,2,3,4,5,6,7,8,9,10,11,12,13,14,15);
}

// gfx1250 async copy: 16B global -> LDS, tracked by ASYNCcnt (no VGPR data).
// Generic pointers to __shared__ hold the LDS byte offset in their low 32 bits.
__device__ __forceinline__ void async_copy_b128(void* lds_ptr, const void* gptr) {
    unsigned int lds_off = (unsigned int)(size_t)lds_ptr;
    unsigned long long ga = (unsigned long long)(size_t)gptr;
    asm volatile("global_load_async_to_lds_b128 %0, %1, off"
                 :: "v"(lds_off), "v"(ga) : "memory");
}
template <int N>
__device__ __forceinline__ void wait_async() {
    asm volatile("s_wait_asynccnt %0" :: "i"(N) : "memory");
}

// ---------------------------------------------------------------------------
// Kernel 1: QKV = x @ w_qkv   ([8192,512] x [512,1536]) -> bf16 Q/K/Vt in ws
// Tile: 128x64 per block; 8 waves in 4x2; each wave 32x32 (2x2 WMMA frags).
// ---------------------------------------------------------------------------
__global__ __launch_bounds__(256) void qkv_gemm(
    const float* __restrict__ x, const float* __restrict__ wqkv,
    bf16* __restrict__ qws, bf16* __restrict__ kws, bf16* __restrict__ vtws) {
    __shared__ bf16 lx[128 * 40];   // X tile 128x32 (stride 40, keeps 16B align)
    __shared__ bf16 lw[64 * 40];    // W tile transposed: [n][k] 64x32

    const int tid  = threadIdx.x;
    const int wv   = tid >> 5;
    const int lane = tid & 31;
    const int wm   = wv & 3, wn = wv >> 2;
    const int row0 = blockIdx.x * 128;
    const int col0 = blockIdx.y * 64;

    const int k2 = (tid >> 4) * 2;   // even k row for W pair-staging
    const int wc4 = tid & 15;        // n-quad for W staging

    v8f acc[2][2] = {};

    for (int kt = 0; kt < CC; kt += 32) {
        // ---- issue ALL staging loads first (one loadcnt wait instead of many)
        float4 vx[4];
        #pragma unroll
        for (int j = 0; j < 4; ++j) {
            int idx = tid + j * 256;
            int r = idx >> 3, c4 = idx & 7;
            vx[j] = *(const float4*)(x + (size_t)(row0 + r) * CC + kt + c4 * 4);
        }
        const float* wp = wqkv + (size_t)(kt + k2) * (3 * CC) + col0 + wc4 * 4;
        float4 w0 = *(const float4*)(wp);
        float4 w1 = *(const float4*)(wp + 3 * CC);

        if (kt + 32 < CC) { // gfx1250 global_prefetch_b8 of next K-step
            __builtin_prefetch(x + (size_t)(row0 + (tid >> 1)) * CC + kt + 32, 0, 1);
            __builtin_prefetch(wqkv + (size_t)(kt + 32 + lane) * (3 * CC) + col0, 0, 1);
        }

        // ---- convert + LDS store
        #pragma unroll
        for (int j = 0; j < 4; ++j) {
            int idx = tid + j * 256;
            int r = idx >> 3, c4 = idx & 7;
            bf16* d = &lx[r * 40 + c4 * 4];
            d[0] = (bf16)vx[j].x; d[1] = (bf16)vx[j].y;
            d[2] = (bf16)vx[j].z; d[3] = (bf16)vx[j].w;
        }
        // W transposed: pack pairs along k -> ds_store_b32
        {
            v2bf t0 = {(bf16)w0.x, (bf16)w1.x};
            v2bf t1 = {(bf16)w0.y, (bf16)w1.y};
            v2bf t2 = {(bf16)w0.z, (bf16)w1.z};
            v2bf t3 = {(bf16)w0.w, (bf16)w1.w};
            *(v2bf*)&lw[(wc4 * 4 + 0) * 40 + k2] = t0;
            *(v2bf*)&lw[(wc4 * 4 + 1) * 40 + k2] = t1;
            *(v2bf*)&lw[(wc4 * 4 + 2) * 40 + k2] = t2;
            *(v2bf*)&lw[(wc4 * 4 + 3) * 40 + k2] = t3;
        }
        __syncthreads();

        #pragma unroll
        for (int fm = 0; fm < 2; ++fm) {
            v16bf a = load_a_frag(&lx[(wm * 32 + fm * 16) * 40], 40);
            #pragma unroll
            for (int fn = 0; fn < 2; ++fn) {
                v16bf b = load_b_frag(&lw[(wn * 32 + fn * 16) * 40], 40);
                acc[fm][fn] = wmma_bf16(a, b, acc[fm][fn]);
            }
        }
        __syncthreads();
    }

    // Epilogue: scatter to Q[b,h,n,d], K[b,h,n,d], Vt[b,h,d,n] (bf16)
    const int cn = lane & 15;
    const int rb = (lane >> 4) * 8;
    #pragma unroll
    for (int fm = 0; fm < 2; ++fm)
        #pragma unroll
        for (int fn = 0; fn < 2; ++fn)
            #pragma unroll
            for (int i = 0; i < 8; ++i) {
                int r = row0 + wm * 32 + fm * 16 + rb + i;
                int c = col0 + wn * 32 + fn * 16 + cn;
                int j = c >> 9, h = (c >> 6) & 7, d = c & 63;
                int bb = r >> 12, n = r & (NN - 1);
                bf16 v = (bf16)acc[fm][fn][i];
                size_t bh = (size_t)(bb * HH + h);
                if (j == 0)      qws[(bh * NN + n) * DD + d] = v;
                else if (j == 1) kws[(bh * NN + n) * DD + d] = v;
                else             vtws[(bh * DD + d) * NN + n] = v;
            }
}

// ---------------------------------------------------------------------------
// Kernel 2: flash attention per (b,h). Block = 128 queries (8 waves x 16).
// K/V tiles double-buffered through LDS with GLOBAL_LOAD_ASYNC_TO_LDS_B128:
// next tile's 4 async copies are in flight while the current tile computes;
// s_wait_asynccnt 4 retires the older group (async completions are in-order).
// ---------------------------------------------------------------------------
__global__ __launch_bounds__(256) void flash_attn(
    const bf16* __restrict__ qws, const bf16* __restrict__ kws,
    const bf16* __restrict__ vtws, bf16* __restrict__ ows) {
    __shared__ bf16 lk[2][64 * 64];     // K tile: [key][d]
    __shared__ bf16 lv[2][64 * 64];     // Vt tile: [d][key]
    __shared__ bf16 lp[8 * 16 * 64];    // per-wave P patch: [q][key]

    const int tid = threadIdx.x, wv = tid >> 5, lane = tid & 31;
    const int h = blockIdx.y, bb = blockIdx.z;
    const int q0 = blockIdx.x * 128 + wv * 16;
    const size_t bh = (size_t)(bb * HH + h);
    const bf16* Q = qws + bh * (size_t)(NN * DD);
    const bf16* K = kws + bh * (size_t)(NN * DD);
    const bf16* V = vtws + bh * (size_t)(DD * NN);

    const int cn = lane & 15;
    const int rb = (lane >> 4) * 8;

    // per-thread staging coordinates: 16B chunks, 2 per thread per tensor
    const int sr0 = tid >> 3,         sc0 = (tid & 7) * 8;
    const int sr1 = (tid + 256) >> 3, sc1 = sc0;   // idx & 7 identical

    // prologue: stage tile 0 into buffer 0
    {
        async_copy_b128(&lk[0][sr0 * 64 + sc0], K + (size_t)sr0 * DD + sc0);
        async_copy_b128(&lv[0][sr0 * 64 + sc0], V + (size_t)sr0 * NN + sc0);
        async_copy_b128(&lk[0][sr1 * 64 + sc1], K + (size_t)sr1 * DD + sc1);
        async_copy_b128(&lv[0][sr1 * 64 + sc1], V + (size_t)sr1 * NN + sc1);
    }

    // Q fragments (2 x 16x32), pre-scaled by D^-0.5 = 0.125 (exact in bf16)
    v16bf qa[2];
    {
        const int row = lane & 15, hlf = lane >> 4;
        #pragma unroll
        for (int dc = 0; dc < 2; ++dc) {
            const bf16* p = Q + (size_t)(q0 + row) * DD + dc * 32 + 8 * hlf;
            v8bf lo = *(const v8bf*)(p);
            v8bf hi = *(const v8bf*)(p + 16);
            v16bf t;
            #pragma unroll
            for (int i = 0; i < 8; ++i) t[i]     = (bf16)((float)lo[i] * 0.125f);
            #pragma unroll
            for (int i = 0; i < 8; ++i) t[8 + i] = (bf16)((float)hi[i] * 0.125f);
            qa[dc] = t;
        }
    }

    v8f o[4] = {};
    float row_m[8], row_l[8];
    #pragma unroll
    for (int i = 0; i < 8; ++i) { row_m[i] = -3.0e38f; row_l[i] = 0.0f; }

    for (int t0 = 0; t0 < NN; t0 += 64) {
        const int buf = (t0 >> 6) & 1;

        // stage next tile into the other buffer, then retire the current group
        if (t0 + 64 < NN) {
            const int nb = buf ^ 1;
            const int nt = t0 + 64;
            async_copy_b128(&lk[nb][sr0 * 64 + sc0], K + (size_t)(nt + sr0) * DD + sc0);
            async_copy_b128(&lv[nb][sr0 * 64 + sc0], V + (size_t)sr0 * NN + nt + sc0);
            async_copy_b128(&lk[nb][sr1 * 64 + sc1], K + (size_t)(nt + sr1) * DD + sc1);
            async_copy_b128(&lv[nb][sr1 * 64 + sc1], V + (size_t)sr1 * NN + nt + sc1);
            wait_async<4>();   // older group of 4 (current tile) has landed
        } else {
            wait_async<0>();
        }
        __syncthreads();

        const bf16* lkb = lk[buf];
        const bf16* lvb = lv[buf];

        // S(16x64) = Qs @ K^T : 4 key-subtiles x 2 d-chunks = 8 WMMA
        v8f s[4];
        #pragma unroll
        for (int ks = 0; ks < 4; ++ks) {
            v8f z = {};
            z = wmma_bf16(qa[0], load_b_frag(&lkb[(ks * 16) * 64 + 0], 64), z);
            z = wmma_bf16(qa[1], load_b_frag(&lkb[(ks * 16) * 64 + 32], 64), z);
            s[ks] = z;
        }

        // online softmax: row lives in one 16-lane half (element i -> row i+8*half)
        float corr[8];
        #pragma unroll
        for (int i = 0; i < 8; ++i) {
            float t = fmaxf(fmaxf(s[0][i], s[1][i]), fmaxf(s[2][i], s[3][i]));
            #pragma unroll
            for (int m = 8; m >= 1; m >>= 1) t = fmaxf(t, __shfl_xor(t, m, 16));
            float mn = fmaxf(row_m[i], t);
            corr[i] = __expf(row_m[i] - mn);
            row_m[i] = mn;
            row_l[i] *= corr[i];
        }
        #pragma unroll
        for (int ks = 0; ks < 4; ++ks)
            #pragma unroll
            for (int i = 0; i < 8; ++i)
                s[ks][i] = __expf(s[ks][i] - row_m[i]);
        #pragma unroll
        for (int i = 0; i < 8; ++i) {
            float t = s[0][i] + s[1][i] + s[2][i] + s[3][i];
            #pragma unroll
            for (int m = 8; m >= 1; m >>= 1) t += __shfl_xor(t, m, 16);
            row_l[i] += t;
        }
        #pragma unroll
        for (int nf = 0; nf < 4; ++nf)
            #pragma unroll
            for (int i = 0; i < 8; ++i) o[nf][i] *= corr[i];

        // re-swizzle P: C-layout -> LDS -> A-layout (per-wave patch, HW in-order)
        bf16* pw = &lp[wv * 16 * 64];
        #pragma unroll
        for (int ks = 0; ks < 4; ++ks)
            #pragma unroll
            for (int i = 0; i < 8; ++i)
                pw[(rb + i) * 64 + ks * 16 + cn] = (bf16)s[ks][i];
        asm volatile("" ::: "memory");  // keep DS store->load order at compile level

        v16bf pa0 = load_a_frag(pw + 0, 64);
        v16bf pa1 = load_a_frag(pw + 32, 64);

        // O(16x64) += P @ Vtile : 4 d-frags x 2 key-chunks = 8 WMMA
        #pragma unroll
        for (int nf = 0; nf < 4; ++nf) {
            o[nf] = wmma_bf16(pa0, load_b_frag(&lvb[(nf * 16) * 64 + 0], 64), o[nf]);
            o[nf] = wmma_bf16(pa1, load_b_frag(&lvb[(nf * 16) * 64 + 32], 64), o[nf]);
        }
        __syncthreads();   // release this buffer for the tile after next
    }

    // normalize and store O as [B, N, H*D] bf16 (matches transpose+reshape)
    float inv[8];
    #pragma unroll
    for (int i = 0; i < 8; ++i) inv[i] = 1.0f / row_l[i];
    #pragma unroll
    for (int nf = 0; nf < 4; ++nf)
        #pragma unroll
        for (int i = 0; i < 8; ++i) {
            int q = q0 + rb + i;
            int d = nf * 16 + cn;
            ows[((size_t)(bb * NN + q)) * CC + h * DD + d] = (bf16)(o[nf][i] * inv[i]);
        }
}

// ---------------------------------------------------------------------------
// Kernel 3: out = O @ w_proj + b_proj  ([8192,512] x [512,512]) -> fp32
// ---------------------------------------------------------------------------
__global__ __launch_bounds__(256) void proj_gemm(
    const bf16* __restrict__ ows, const float* __restrict__ wproj,
    const float* __restrict__ bproj, float* __restrict__ out) {
    __shared__ bf16 la[128 * 40];
    __shared__ bf16 lw[64 * 40];

    const int tid  = threadIdx.x;
    const int wv   = tid >> 5;
    const int lane = tid & 31;
    const int wm   = wv & 3, wn = wv >> 2;
    const int row0 = blockIdx.x * 128;
    const int col0 = blockIdx.y * 64;

    const int k2 = (tid >> 4) * 2;
    const int wc4 = tid & 15;

    v8f acc[2][2] = {};

    for (int kt = 0; kt < CC; kt += 32) {
        // ---- issue all staging loads first
        v8bf va[2];
        #pragma unroll
        for (int j = 0; j < 2; ++j) {
            int idx = tid + j * 256;
            int r = idx >> 2, c8 = idx & 3;
            va[j] = *(const v8bf*)(ows + (size_t)(row0 + r) * CC + kt + c8 * 8);
        }
        const float* wp = wproj + (size_t)(kt + k2) * CC + col0 + wc4 * 4;
        float4 w0 = *(const float4*)(wp);
        float4 w1 = *(const float4*)(wp + CC);

        if (kt + 32 < CC) {
            __builtin_prefetch(ows + (size_t)(row0 + (tid >> 1)) * CC + kt + 32, 0, 1);
            __builtin_prefetch(wproj + (size_t)(kt + 32 + lane) * CC + col0, 0, 1);
        }

        // ---- store
        #pragma unroll
        for (int j = 0; j < 2; ++j) {
            int idx = tid + j * 256;
            int r = idx >> 2, c8 = idx & 3;
            *(v8bf*)&la[r * 40 + c8 * 8] = va[j];
        }
        {
            v2bf t0 = {(bf16)w0.x, (bf16)w1.x};
            v2bf t1 = {(bf16)w0.y, (bf16)w1.y};
            v2bf t2 = {(bf16)w0.z, (bf16)w1.z};
            v2bf t3 = {(bf16)w0.w, (bf16)w1.w};
            *(v2bf*)&lw[(wc4 * 4 + 0) * 40 + k2] = t0;
            *(v2bf*)&lw[(wc4 * 4 + 1) * 40 + k2] = t1;
            *(v2bf*)&lw[(wc4 * 4 + 2) * 40 + k2] = t2;
            *(v2bf*)&lw[(wc4 * 4 + 3) * 40 + k2] = t3;
        }
        __syncthreads();

        #pragma unroll
        for (int fm = 0; fm < 2; ++fm) {
            v16bf a = load_a_frag(&la[(wm * 32 + fm * 16) * 40], 40);
            #pragma unroll
            for (int fn = 0; fn < 2; ++fn) {
                v16bf b = load_b_frag(&lw[(wn * 32 + fn * 16) * 40], 40);
                acc[fm][fn] = wmma_bf16(a, b, acc[fm][fn]);
            }
        }
        __syncthreads();
    }

    const int cn = lane & 15;
    const int rb = (lane >> 4) * 8;
    #pragma unroll
    for (int fm = 0; fm < 2; ++fm)
        #pragma unroll
        for (int fn = 0; fn < 2; ++fn)
            #pragma unroll
            for (int i = 0; i < 8; ++i) {
                int r = row0 + wm * 32 + fm * 16 + rb + i;
                int c = col0 + wn * 32 + fn * 16 + cn;
                out[(size_t)r * CC + c] = acc[fm][fn][i] + bproj[c];
            }
}

// ---------------------------------------------------------------------------
extern "C" void kernel_launch(void* const* d_in, const int* in_sizes, int n_in,
                              void* d_out, int out_size, void* d_ws, size_t ws_size,
                              hipStream_t stream) {
    (void)in_sizes; (void)n_in; (void)out_size; (void)ws_size;
    const float* x     = (const float*)d_in[0];
    const float* wqkv  = (const float*)d_in[1];
    const float* wproj = (const float*)d_in[2];
    const float* bproj = (const float*)d_in[3];
    float* out = (float*)d_out;

    const size_t per = (size_t)BB * HH * NN * DD;  // 4,194,304 elems (8 MiB bf16)
    bf16* qws = (bf16*)d_ws;
    bf16* kws = qws + per;
    bf16* vt  = kws + per;
    bf16* ows = vt + per;      // [B, N, C] attention output, bf16

    // 1) QKV projection: [8192,512] x [512,1536]
    qkv_gemm<<<dim3((BB * NN) / 128, (3 * CC) / 64), 256, 0, stream>>>(
        x, wqkv, qws, kws, vt);
    // 2) flash attention: 32 query-blocks x H x B
    flash_attn<<<dim3(NN / 128, HH, BB), 256, 0, stream>>>(qws, kws, vt, ows);
    // 3) output projection + bias: [8192,512] x [512,512]
    proj_gemm<<<dim3((BB * NN) / 128, CC / 64), 256, 0, stream>>>(
        ows, wproj, bproj, out);
}
